// VarianceLoss_73727408603749
// MI455X (gfx1250) — matrix operand order repaired and checked by reading, not compiled, not run on hardware
//
#include <hip/hip_runtime.h>
#include <hip/hip_bf16.h>
#include <math.h>

typedef __attribute__((ext_vector_type(16))) _Float16 v16h;
typedef __attribute__((ext_vector_type(8)))  float    v8f;

#define BB    8
#define NN    200
#define KK    32
#define HP    138
#define PP    (HP * HP)        // 19044 pixels
#define NPAD  208              // N padded to 13 * 16
#define NCH   13               // detection chunks of 16
#define H0    550
#define TILES ((PP + 15) / 16) // 1191 pixel tiles of 16
#define PPAD  (TILES * 16)     // 19056

// ---------------------------------------------------------------------------
// Kernel 1: separable gaussian tables. gxc = exp(-dx^2/(2 sx^2)) * conf,
// gy = exp(-dy^2/(2 sy^2)). Padded detections (n >= 200) are zero so they
// contribute nothing downstream.
// ---------------------------------------------------------------------------
__global__ void gauss_kernel(const float* __restrict__ loc,
                             const float* __restrict__ conf,
                             float* __restrict__ gxc,
                             float* __restrict__ gy) {
  int idx = blockIdx.x * blockDim.x + threadIdx.x;
  const int total = BB * NPAD * HP;
  if (idx >= total) return;
  int j = idx % HP;
  int t = idx / HP;
  int n = t % NPAD;
  int b = t / NPAD;
  float gxv = 0.0f, gyv = 0.0f;
  if (n < NN) {
    const float* l = loc + ((size_t)(b * NN + n)) * 4;
    float cx = l[0], cy = l[1], w = l[2], h = l[3];
    float cf = conf[b * NN + n];
    float coord = (j + 0.5f) / (float)HP;   // Hp == Wp, same grid both axes
    float sx = fmaxf(w, 1e-3f);
    float sy = fmaxf(h, 1e-3f);
    float dx = coord - cx;
    float dy = coord - cy;
    gxv = __expf(-(dx * dx) / (2.0f * sx * sx)) * cf;
    gyv = __expf(-(dy * dy) / (2.0f * sy * sy));
  }
  gxc[idx] = gxv;
  gy[idx]  = gyv;
}

// ---------------------------------------------------------------------------
// Kernel 1b: pre-pack mask [B,200,32] f32 into WMMA B-matrix f16 lane layout:
// maskB[(b*NCH+ch)*32 + lane][0..15] = the v16h fragment lane needs for chunk
// ch (16-bit B layout: VGPR i holds K = (i<4 ? 2i : 8+2i) + 8*(lane>>4), +1).
// ---------------------------------------------------------------------------
__global__ void pack_maskB_kernel(const float* __restrict__ maskm,
                                  _Float16* __restrict__ maskB) {
  int idx = blockIdx.x * blockDim.x + threadIdx.x;  // (b*NCH + ch)*32 + lane
  if (idx >= BB * NCH * 32) return;
  int lane = idx & 31;
  int t    = idx >> 5;
  int ch   = t % NCH;
  int b    = t / NCH;
  int half = lane >> 4;
  int col  = lane & 15;
  int n    = ch * 16 + col;
  int nld  = (n < NN) ? n : (NN - 1);   // clamp; gxc==0 kills padded columns
  const float* pm = maskm + ((size_t)b * NN + nld) * KK;
  _Float16* dst = maskB + (size_t)idx * 16;
#pragma unroll
  for (int i = 0; i < 8; ++i) {
    int kB = (i < 4 ? 2 * i : 8 + 2 * i) + half * 8;
    dst[2 * i]     = (_Float16)pm[kB];
    dst[2 * i + 1] = (_Float16)pm[kB + 1];
  }
}

// ---------------------------------------------------------------------------
// Kernel 2: fused  sigmoid(proto @ mask^T) * gauss * conf  with streaming
// reduction over detections. One wave32 per (batch, 16-pixel tile).
//   A (16x32 f16) = proto rows, loaded once per wave, reused for 13 chunks.
//   B (32x16 f16) = pre-packed fragment, one 32B v16h load per lane per chunk,
//                   software-pipelined one chunk ahead.
//   D layout: VGPR r, lane l -> pixel = tile*16 + r + 8*(l>>4), det col = l&15.
// Epilogue is branchless; gaussian gathers are batched so the compiler can
// issue them as one clause with a single wait.
// ---------------------------------------------------------------------------
__global__ void wmma_conf_kernel(const float* __restrict__ proto,
                                 const _Float16* __restrict__ maskB,
                                 const float* __restrict__ gxc,
                                 const float* __restrict__ gy,
                                 float* __restrict__ fc) {
  int g    = blockIdx.x * (blockDim.x >> 5) + (threadIdx.x >> 5);
  int lane = threadIdx.x & 31;
  int b    = g / TILES;
  int tile = g - b * TILES;
  int half = lane >> 4;   // 0: lanes 0-15, 1: lanes 16-31
  int col  = lane & 15;   // detection column within chunk / pixel row for A

  // ---- Load A matrix: 16 pixels x K=32 of proto, f32 -> f16 (once) ----
  // 16-bit A layout: VGPR pair i holds K = (i<4 ? 2i : 8+2i) + 8*half, +1
  int pixA = tile * 16 + col;
  if (pixA >= PP) pixA = PP - 1;  // clamp; padded rows never stored
  const float* pa = proto + ((size_t)b * PP + pixA) * KK;
  v16h a;
#pragma unroll
  for (int i = 0; i < 8; ++i) {
    int kA = (i < 4 ? 2 * i : 8 + 2 * i) + half * 8;
    a[2 * i]     = (_Float16)pa[kA];
    a[2 * i + 1] = (_Float16)pa[kA + 1];
  }

  // ---- Hoist per-r pixel coordinates (independent of chunk) ----
  int wofs[8];
  bool topRow[8];
  int h0 = (tile * 16) / HP;
  int lastpix = tile * 16 + 15;
  if (lastpix >= PP) lastpix = PP - 1;
  int h1 = lastpix / HP;             // tile spans at most 2 rows
#pragma unroll
  for (int r = 0; r < 8; ++r) {
    int pix = tile * 16 + r + half * 8;
    if (pix >= PP) pix = PP - 1;     // branchless clamp; store is guarded
    int hh = pix / HP;
    wofs[r]   = pix - hh * HP;
    topRow[r] = (hh == h0);
  }

  float s1[8], s2[8];
#pragma unroll
  for (int r = 0; r < 8; ++r) { s1[r] = 0.0f; s2[r] = 0.0f; }

  // Walking pointers: column col of chunk ch is detection n = ch*16 + col.
  const float*    gx_n = gxc   + ((size_t)b * NPAD + col) * HP;
  const float*    gy_n = gy    + ((size_t)b * NPAD + col) * HP;
  const _Float16* bptr = maskB + (((size_t)b * NCH) * 32 + lane) * 16;

  v16h bm = *(const v16h*)bptr;      // prologue: fragment for chunk 0

  for (int ch = 0; ch < NCH; ++ch) {
    v8f c = {};
    c = __builtin_amdgcn_wmma_f32_16x16x32_f16(
        /*neg_a=*/false, a, /*neg_b=*/false, bm,
        /*c_mod=*/(short)0, c, /*reuse_a=*/false, /*reuse_b=*/false);

    // Pipeline: fetch next chunk's fragment under the epilogue math.
    v16h bm_next = bm;
    if (ch + 1 < NCH) bm_next = *(const v16h*)(bptr + 32 * 16);

    // Batched gaussian gathers: issue all loads, then one wait, then math.
    float gxv[8];
#pragma unroll
    for (int r = 0; r < 8; ++r) gxv[r] = gx_n[wofs[r]];
    float gy0 = gy_n[h0];
    float gy1 = gy_n[h1];

#pragma unroll
    for (int r = 0; r < 8; ++r) {
      float x   = c[r];
      float sg  = __builtin_amdgcn_rcpf(1.0f + __expf(-x));  // fast sigmoid
      float gyv = topRow[r] ? gy0 : gy1;
      float v   = sg * gxv[r] * gyv;  // * gauss * conf
      s1[r] += v;
      s2[r] += v * v;
    }
    gx_n += 16 * HP;
    gy_n += 16 * HP;
    bptr += 32 * 16;
    bm = bm_next;
  }

  // Reduce over the 16 detection columns (xor masks 1,2,4,8 stay in-half).
#pragma unroll
  for (int r = 0; r < 8; ++r) {
#pragma unroll
    for (int m = 8; m >= 1; m >>= 1) {
      s1[r] += __shfl_xor(s1[r], m, 32);
      s2[r] += __shfl_xor(s2[r], m, 32);
    }
  }
  if (col == 0) {
#pragma unroll
    for (int r = 0; r < 8; ++r) {
      int pix = tile * 16 + r + half * 8;
      float fv = 1.0f - s2[r] / (s1[r] + 1e-6f);
      if (fv != fv) fv = 0.0f;  // NaN -> 0
      if (pix < PP) fc[(size_t)b * PPAD + pix] = fv;
    }
  }
}

// ---------------------------------------------------------------------------
// Kernel 3: bilinear resize 138 -> 550 (half-pixel centers, edge clamp) fused
// with weighted mean / variance over batch; per-block partial sums.
// ---------------------------------------------------------------------------
__global__ void resize_var_kernel(const float* __restrict__ orig,
                                  const float* __restrict__ fc,
                                  float* __restrict__ partials) {
  int idx = blockIdx.x * blockDim.x + threadIdx.x;
  float acc = 0.0f;
  if (idx < H0 * H0) {
    int x = idx % H0;
    int y = idx / H0;
    const float scale = (float)HP / (float)H0;
    float sxf = (x + 0.5f) * scale - 0.5f;
    float syf = (y + 0.5f) * scale - 0.5f;
    float fx0 = floorf(sxf), fy0 = floorf(syf);
    float tx = sxf - fx0, ty = syf - fy0;
    int x0 = (int)fx0, y0 = (int)fy0;
    int x1 = x0 + 1, y1 = y0 + 1;
    x0 = min(max(x0, 0), HP - 1); x1 = min(max(x1, 0), HP - 1);
    y0 = min(max(y0, 0), HP - 1); y1 = min(max(y1, 0), HP - 1);

    float rs[BB];
    float total = 0.0f;
#pragma unroll
    for (int b = 0; b < BB; ++b) {
      const float* f = fc + (size_t)b * PPAD;
      float v00 = f[y0 * HP + x0], v01 = f[y0 * HP + x1];
      float v10 = f[y1 * HP + x0], v11 = f[y1 * HP + x1];
      float v0 = v00 + (v01 - v00) * tx;
      float v1 = v10 + (v11 - v10) * tx;
      float r  = v0 + (v1 - v0) * ty;
      rs[b] = r;
      total += r;
    }
#pragma unroll
    for (int c = 0; c < 3; ++c) {
      float wm = 0.0f;
      float ov[BB];
#pragma unroll
      for (int b = 0; b < BB; ++b) {
        float o = orig[(((size_t)b * 3 + c) * H0 + y) * H0 + x];
        ov[b] = o;
        wm += o * rs[b];
      }
      wm /= total;  // weighted_mean: no epsilon (matches reference)
#pragma unroll
      for (int b = 0; b < BB; ++b) {
        float d = ov[b] - wm;
        acc += d * d * rs[b];
      }
    }
    acc /= (total + 1e-6f);  // weighted_var epsilon
  }

  __shared__ float red[256];
  red[threadIdx.x] = acc;
  __syncthreads();
  for (int s = 128; s > 0; s >>= 1) {
    if ((int)threadIdx.x < s) red[threadIdx.x] += red[threadIdx.x + s];
    __syncthreads();
  }
  if (threadIdx.x == 0) partials[blockIdx.x] = red[0];
}

// ---------------------------------------------------------------------------
// Kernel 4: deterministic final reduction, scale by B/K = 0.25.
// ---------------------------------------------------------------------------
__global__ void final_reduce_kernel(const float* __restrict__ partials, int n,
                                    float* __restrict__ out) {
  __shared__ float red[256];
  float a = 0.0f;
  for (int i = threadIdx.x; i < n; i += 256) a += partials[i];
  red[threadIdx.x] = a;
  __syncthreads();
  for (int s = 128; s > 0; s >>= 1) {
    if ((int)threadIdx.x < s) red[threadIdx.x] += red[threadIdx.x + s];
    __syncthreads();
  }
  if (threadIdx.x == 0) out[0] = red[0] * ((float)BB / (float)KK);
}

// ---------------------------------------------------------------------------
extern "C" void kernel_launch(void* const* d_in, const int* in_sizes, int n_in,
                              void* d_out, int out_size, void* d_ws,
                              size_t ws_size, hipStream_t stream) {
  const float* orig  = (const float*)d_in[0];  // [8,3,550,550]
  const float* loc   = (const float*)d_in[1];  // [8,200,4]
  const float* maskm = (const float*)d_in[2];  // [8,200,32]
  const float* conf  = (const float*)d_in[3];  // [8,200]
  const float* proto = (const float*)d_in[4];  // [8,138,138,32]
  float* out = (float*)d_out;

  float* ws  = (float*)d_ws;
  float* gxc = ws;                                   // 229632 floats
  float* gy  = gxc + (size_t)BB * NPAD * HP;         // 229632 floats
  float* fc  = gy  + (size_t)BB * NPAD * HP;         // 152448 floats
  float* partials = fc + (size_t)BB * PPAD;          // 1182 floats
  // 128-byte aligned f16 region for packed WMMA-B fragments (53248 halfs)
  _Float16* maskB = (_Float16*)(ws + 612928);

  int totA = BB * NPAD * HP;
  gauss_kernel<<<(totA + 255) / 256, 256, 0, stream>>>(loc, conf, gxc, gy);

  int totP = BB * NCH * 32;  // 3328
  pack_maskB_kernel<<<(totP + 255) / 256, 256, 0, stream>>>(maskm, maskB);

  // 8 waves/block, one wave per (b, tile): TILES blocks cover 8*TILES waves.
  wmma_conf_kernel<<<TILES, 256, 0, stream>>>(proto, maskB, gxc, gy, fc);

  int nPix = H0 * H0;
  int nBlocksC = (nPix + 255) / 256;  // 1182
  resize_var_kernel<<<nBlocksC, 256, 0, stream>>>(orig, fc, partials);

  final_reduce_kernel<<<1, 256, 0, stream>>>(partials, nBlocksC, out);
}
